// DualStreamSDENet_67860483277138
// MI455X (gfx1250) — compile-verified, matrix operand order checked by reading, not run on hardware
//
#include <hip/hip_runtime.h>
#include <hip/hip_bf16.h>
#include <math.h>

typedef __attribute__((ext_vector_type(16))) _Float16 v16h;
typedef __attribute__((ext_vector_type(8)))  float    v8f;

#define BATCH   16384
#define TT      64
#define DD      64
#define HH      128
#define AUXN    10
#define MIDN    100
#define MIDP    112
#define DEPTH_N 25
#define FSK     96      // padded D+AUX (74 -> 96)
#define FSK_REAL 74

// f16 weight workspace layout (offsets in halves)
#define OFF_DS   0        // 128x64
#define OFF_DR   8192     // 128x128
#define OFF_DF1  24576    // 112x128 (rows 100..111 zero)
#define OFF_FS   38912    // 128x96  (cols 74..95 zero)
#define OFF_IN   51200    // 384x128
#define OFF_OUT  100352   // 128x128
#define OFF_TR   116736   // 128x256
#define WS_HALVES 149504

// -------------------- weight f32 -> f16 pre-pass --------------------
__global__ void cvt_weights_kernel(const float* ds_w, const float* dr_w,
                                   const float* df_w1, const float* fs_w,
                                   const float* in_w, const float* out_w,
                                   const float* tr_w, _Float16* ws) {
  int i = blockIdx.x * 256 + threadIdx.x;
  if (i >= WS_HALVES) return;
  float v;
  if (i < OFF_DR)        { v = ds_w[i - OFF_DS]; }
  else if (i < OFF_DF1)  { v = dr_w[i - OFF_DR]; }
  else if (i < OFF_FS)   { int j = i - OFF_DF1; int r = j >> 7, c = j & 127;
                           v = (r < MIDN) ? df_w1[r * HH + c] : 0.0f; }
  else if (i < OFF_IN)   { int j = i - OFF_FS; int r = j / FSK, c = j % FSK;
                           v = (c < FSK_REAL) ? fs_w[r * FSK_REAL + c] : 0.0f; }
  else if (i < OFF_OUT)  { v = in_w[i - OFF_IN]; }
  else if (i < OFF_TR)   { v = out_w[i - OFF_OUT]; }
  else                   { v = tr_w[i - OFF_TR]; }
  ws[i] = (_Float16)v;
}

// -------------------- WMMA fragment helpers (ISA 7.12.2 layouts) --------------------
// A (16x32, f16): lanes 0-15 row M=lane, halves 0..7 = K 0..7, 8..15 = K 16..23;
//                 lanes 16-31 row M=lane-16, halves 0..7 = K 8..15, 8..15 = K 24..31.
__device__ __forceinline__ v16h a_from_f32(const float* base, int ld, int k0, int lane) {
  int m = lane & 15, kb = (lane >> 4) * 8;
  const float* p = base + (size_t)m * ld + k0 + kb;
  v16h a;
#pragma unroll
  for (int i = 0; i < 8; ++i) a[i] = (_Float16)p[i];
#pragma unroll
  for (int i = 0; i < 8; ++i) a[8 + i] = (_Float16)p[16 + i];
  return a;
}
// B = W^T tile: lane n holds column (n0+n) of W^T == row (n0+n) of W; K packing as A.
__device__ __forceinline__ v16h b_from_f16(const _Float16* W, int ld, int n0, int k0, int lane) {
  int n = lane & 15, kb = (lane >> 4) * 8;
  const _Float16* p = W + (size_t)(n0 + n) * ld + k0 + kb;
  v16h b;
#pragma unroll
  for (int i = 0; i < 8; ++i) b[i] = p[i];
#pragma unroll
  for (int i = 0; i < 8; ++i) b[8 + i] = p[16 + i];
  return b;
}
// C/D: lane = N (lo/hi half), VGPR j -> M = j + 8*(lane>=16)
__device__ __forceinline__ void store_c_f32(float* dst, int ld, int n0, int lane, v8f c) {
  int col = n0 + (lane & 15), rb = (lane >> 4) * 8;
#pragma unroll
  for (int j = 0; j < 8; ++j) dst[(size_t)(rb + j) * ld + col] = c[j];
}
__device__ __forceinline__ v8f bias8(const float* bias, int col) {
  float v = bias[col];
  v8f c;
#pragma unroll
  for (int j = 0; j < 8; ++j) c[j] = v;
  return c;
}
__device__ __forceinline__ v8f bias8pad(const float* bias, int col, int n) {
  float v = (col < n) ? bias[col] : 0.0f;
  v8f c;
#pragma unroll
  for (int j = 0; j < 8; ++j) c[j] = v;
  return c;
}
__device__ __forceinline__ v8f wmma16(v16h a, v16h b, v8f c) {
  return __builtin_amdgcn_wmma_f32_16x16x32_f16(false, a, false, b, (short)0, c, false, false);
}

struct Params {
  const float *x_seq, *aux, *levy;
  const float *ds_b, *dr_b, *df_b1, *df_w2, *df_b2;
  const float *g_w1, *g_b1, *g_w2, *g_b2;
  const float *fs_b, *in_b, *out_b, *ln_g, *ln_b, *tr_b;
  const float *hm_w, *hm_b, *hs_w, *hs_b, *hd_w, *hd_b;
  const _Float16* wh;
  float *o_mean, *o_ls, *o_dir;
};

#define WAVE_LDS 57408   // 8192 (h) + 8192 (feat) + 40960 (scratch) + 64 (nc)
#define WG_LDS   (32768 + 4 * WAVE_LDS)

// -------------------- fused forward kernel: 4 waves/WG, 16 rows/wave --------------------
__global__ void __launch_bounds__(128)
DualStreamSDENet_fused_kernel(Params P) {
  extern __shared__ char smem[];
  const int tid  = threadIdx.x;
  const int w    = tid >> 5;
  const int lane = tid & 31;
  const int b0   = blockIdx.x * 64 + w * 16;   // 16 batch rows per wave
  const int rb   = (lane >> 4) * 8;            // C-fragment row base for this lane
  const int nlo  = lane & 15;

  _Float16* dr16 = (_Float16*)smem;                    // [128][128] f16, shared by WG
  char* wbase = smem + 32768 + (size_t)w * WAVE_LDS;
  float*    hbuf  = (float*)wbase;                     // [16][128] f32  (H_lde state)
  float*    fbuf  = (float*)(wbase + 8192);            // [16][128] f32  (H_feat)
  char*     scr   = wbase + 16384;                     // 40960 B overlaid scratch
  float*    ncbuf = (float*)(wbase + 16384 + 40960);   // [16] f32

  // stage dr_w (f16) into LDS: reused 25x by every wave
  {
    const unsigned int* src = (const unsigned int*)(P.wh + OFF_DR);
    unsigned int* dst = (unsigned int*)dr16;
    for (int i = tid; i < 8192; i += 128) dst[i] = src[i];
  }
  __syncthreads();

  // ---------------- Phase A: out = x_last @ ds_w^T + ds_b ----------------
  v8f hc[8];
  {
    const float* xlast = P.x_seq + ((size_t)b0 * TT + (TT - 1)) * DD;
    v16h a0 = a_from_f32(xlast, TT * DD, 0, lane);
    v16h a1 = a_from_f32(xlast, TT * DD, 32, lane);
#pragma unroll
    for (int nt = 0; nt < 8; ++nt) {
      v8f c = bias8(P.ds_b, nt * 16 + nlo);
      c = wmma16(a0, b_from_f16(P.wh + OFF_DS, DD, nt * 16, 0, lane), c);
      c = wmma16(a1, b_from_f16(P.wh + OFF_DS, DD, nt * 16, 32, lane), c);
      hc[nt] = c;
      store_c_f32(hbuf, HH, nt * 16, lane, c);
    }
  }
  __syncthreads();

  // ---------------- Phase B: per-row noise_coef ----------------
  {
    float* mid = (float*)scr;  // [16][112]
    v16h ha[4];
#pragma unroll
    for (int kt = 0; kt < 4; ++kt) ha[kt] = a_from_f32(hbuf, HH, kt * 32, lane);
#pragma unroll
    for (int nt = 0; nt < 7; ++nt) {
      int col = nt * 16 + nlo;
      v8f c = bias8pad(P.df_b1, col, MIDN);
#pragma unroll
      for (int kt = 0; kt < 4; ++kt)
        c = wmma16(ha[kt], b_from_f16(P.wh + OFF_DF1, HH, nt * 16, kt * 32, lane), c);
#pragma unroll
      for (int j = 0; j < 8; ++j) c[j] = fmaxf(c[j], 0.0f);
      store_c_f32(mid, MIDP, nt * 16, lane, c);
    }
    if (lane < 16) {
      int r = lane; size_t b = (size_t)b0 + r;
      float ax[AUXN];
#pragma unroll
      for (int i = 0; i < AUXN; ++i) ax[i] = P.aux[b * AUXN + i];
      float g2 = P.g_b2[0];
#pragma unroll
      for (int mi = 0; mi < 16; ++mi) {
        float t = P.g_b1[mi];
#pragma unroll
        for (int a2 = 0; a2 < AUXN; ++a2) t += P.g_w1[mi * AUXN + a2] * ax[a2];
        g2 += P.g_w2[mi] * fmaxf(t, 0.0f);
      }
      float gate  = 1.0f / (1.0f + expf(-g2));
      float macro = 0.1f + 1.4f * gate;                 // SIGMA_MIN + (MAX-MIN)*gate
      float acc = P.df_b2[0];
      for (int j = 0; j < MIDN; ++j) acc += mid[r * MIDP + j] * P.df_w2[j];
      float micro = 1.0f / (1.0f + expf(-acc));
      ncbuf[r] = macro * micro * powf(0.04f, 1.0f / 1.2f);  // dt^(1/alpha)
    }
  }
  __syncthreads();

  // ---------------- Phase C: 25-step Levy SDE scan (h in WMMA accumulators) ----------------
  {
    float ncr[8];
#pragma unroll
    for (int j = 0; j < 8; ++j) ncr[j] = ncbuf[rb + j];
    const float dt = 0.04f;
    for (int step = 0; step < DEPTH_N; ++step) {
      const float* eta = P.levy + ((size_t)step * BATCH + b0) * HH;
      if (step + 1 < DEPTH_N) {
        const float* nx = P.levy + ((size_t)(step + 1) * BATCH + b0) * HH;
        __builtin_prefetch(nx + lane * 32, 0, 1);          // global_prefetch_b8
        __builtin_prefetch(nx + 1024 + lane * 32, 0, 1);
      }
      v16h a[4];
#pragma unroll
      for (int kt = 0; kt < 4; ++kt) a[kt] = a_from_f32(hbuf, HH, kt * 32, lane);
#pragma unroll
      for (int nt = 0; nt < 8; ++nt) {
        int col = nt * 16 + nlo;
        v8f g = bias8(P.dr_b, col);
#pragma unroll
        for (int kt = 0; kt < 4; ++kt)
          g = wmma16(a[kt], b_from_f16(dr16, HH, nt * 16, kt * 32, lane), g);
        v8f c = hc[nt];
#pragma unroll
        for (int j = 0; j < 8; ++j)
          c[j] = c[j] + fmaxf(g[j], 0.0f) * dt + ncr[j] * eta[(size_t)(rb + j) * HH + col];
        hc[nt] = c;
        store_c_f32(hbuf, HH, nt * 16, lane, c);
      }
    }
  }
  __syncthreads();

  // ---------------- Phase D: x_mean over T, concat aux, pad to 96 ----------------
  {
    float* xm = (float*)scr;  // [16][96]
    int m = lane & 15, hi = lane >> 4;
    for (int r = 0; r < 16; ++r) {
      const float* xr = P.x_seq + ((size_t)(b0 + r) * TT) * DD + 4 * m;
      float4 acc = make_float4(0.f, 0.f, 0.f, 0.f);
      for (int t = hi; t < TT; t += 2) {
        const float4 v = *(const float4*)(xr + (size_t)t * DD);
        acc.x += v.x; acc.y += v.y; acc.z += v.z; acc.w += v.w;
      }
      acc.x += __shfl_xor(acc.x, 16, 32);
      acc.y += __shfl_xor(acc.y, 16, 32);
      acc.z += __shfl_xor(acc.z, 16, 32);
      acc.w += __shfl_xor(acc.w, 16, 32);
      if (hi == 0) {
        const float inv = 1.0f / (float)TT;
        xm[r * FSK + 4 * m + 0] = acc.x * inv;
        xm[r * FSK + 4 * m + 1] = acc.y * inv;
        xm[r * FSK + 4 * m + 2] = acc.z * inv;
        xm[r * FSK + 4 * m + 3] = acc.w * inv;
      }
    }
    if (lane < 16) {
      int r = lane;
      for (int c = 64; c < FSK; ++c)
        xm[r * FSK + c] = (c < FSK_REAL) ? P.aux[(size_t)(b0 + r) * AUXN + (c - 64)] : 0.0f;
    }
  }
  __syncthreads();

  // ---------------- Phase E: H_feat = relu([x_mean,aux] @ fs_w^T + fs_b) ----------------
  {
    const float* xm = (const float*)scr;
    v16h fa[3];
#pragma unroll
    for (int kt = 0; kt < 3; ++kt) fa[kt] = a_from_f32(xm, FSK, kt * 32, lane);
#pragma unroll
    for (int nt = 0; nt < 8; ++nt) {
      v8f c = bias8(P.fs_b, nt * 16 + nlo);
#pragma unroll
      for (int kt = 0; kt < 3; ++kt)
        c = wmma16(fa[kt], b_from_f16(P.wh + OFF_FS, FSK, nt * 16, kt * 32, lane), c);
#pragma unroll
      for (int j = 0; j < 8; ++j) c[j] = fmaxf(c[j], 0.0f);
      store_c_f32(fbuf, HH, nt * 16, lane, c);
    }
  }
  __syncthreads();

  // ---------------- Phase F: qkv, seq-len-2 attention, out-proj, residual, LN ----------------
  _Float16* qkv = (_Float16*)scr;            // [2][16][384] f16
  float*    ctx = (float*)(scr + 24576);     // [2][16][128] f32
  for (int s = 0; s < 2; ++s) {
    const float* src = s ? fbuf : hbuf;
    v16h qa[4];
#pragma unroll
    for (int kt = 0; kt < 4; ++kt) qa[kt] = a_from_f32(src, HH, kt * 32, lane);
#pragma unroll
    for (int nt = 0; nt < 24; ++nt) {
      int col = nt * 16 + nlo;
      v8f c = bias8(P.in_b, col);
#pragma unroll
      for (int kt = 0; kt < 4; ++kt)
        c = wmma16(qa[kt], b_from_f16(P.wh + OFF_IN, HH, nt * 16, kt * 32, lane), c);
#pragma unroll
      for (int j = 0; j < 8; ++j)
        qkv[((size_t)s * 16 + rb + j) * 384 + col] = (_Float16)c[j];
    }
  }
  __syncthreads();
  if (lane < 16) {
    int r = lane;
    const _Float16* q0 = qkv + (size_t)r * 384;
    const _Float16* q1 = qkv + (size_t)(16 + r) * 384;
#pragma unroll
    for (int hd = 0; hd < 4; ++hd) {
      int o = hd * 32;
      float s00 = 0, s01 = 0, s10 = 0, s11 = 0;
      for (int i = 0; i < 32; ++i) {
        float a0 = (float)q0[o + i],       a1 = (float)q1[o + i];
        float k0 = (float)q0[128 + o + i], k1 = (float)q1[128 + o + i];
        s00 += a0 * k0; s01 += a0 * k1; s10 += a1 * k0; s11 += a1 * k1;
      }
      const float is = 0.17677669529663687f;  // 1/sqrt(32)
      s00 *= is; s01 *= is; s10 *= is; s11 *= is;
      float m0 = fmaxf(s00, s01), e0 = expf(s00 - m0), e1 = expf(s01 - m0);
      float a00 = e0 / (e0 + e1), a01 = e1 / (e0 + e1);
      float m1 = fmaxf(s10, s11), f0 = expf(s10 - m1), f1 = expf(s11 - m1);
      float a10 = f0 / (f0 + f1), a11 = f1 / (f0 + f1);
      for (int i = 0; i < 32; ++i) {
        float v0 = (float)q0[256 + o + i], v1 = (float)q1[256 + o + i];
        ctx[(size_t)r * HH + o + i]        = a00 * v0 + a01 * v1;
        ctx[(size_t)(16 + r) * HH + o + i] = a10 * v0 + a11 * v1;
      }
    }
  }
  __syncthreads();
  float* resid = (float*)scr;  // [16][256] f32 (reuses dead qkv bytes)
  for (int s = 0; s < 2; ++s) {
    const float* cs   = ctx + (size_t)s * 16 * HH;
    const float* seqb = s ? fbuf : hbuf;
    v16h ca[4];
#pragma unroll
    for (int kt = 0; kt < 4; ++kt) ca[kt] = a_from_f32(cs, HH, kt * 32, lane);
#pragma unroll
    for (int nt = 0; nt < 8; ++nt) {
      int col = nt * 16 + nlo;
      v8f c = bias8(P.out_b, col);
#pragma unroll
      for (int kt = 0; kt < 4; ++kt)
        c = wmma16(ca[kt], b_from_f16(P.wh + OFF_OUT, HH, nt * 16, kt * 32, lane), c);
#pragma unroll
      for (int j = 0; j < 8; ++j) c[j] += seqb[(size_t)(rb + j) * HH + col];
      store_c_f32(resid, 256, s * 128 + nt * 16, lane, c);
    }
  }
  __syncthreads();
  if (lane < 16) {
    int r = lane;
#pragma unroll
    for (int s = 0; s < 2; ++s) {
      float* row = resid + (size_t)r * 256 + s * 128;
      float mu = 0;
      for (int c = 0; c < HH; ++c) mu += row[c];
      mu *= (1.0f / HH);
      float var = 0;
      for (int c = 0; c < HH; ++c) { float d = row[c] - mu; var += d * d; }
      var *= (1.0f / HH);
      float rstd = rsqrtf(var + 1e-5f);
      for (int c = 0; c < HH; ++c)
        row[c] = (row[c] - mu) * rstd * P.ln_g[c] + P.ln_b[c];
    }
  }
  __syncthreads();

  // ---------------- Phase G: h = relu(fused @ tr_w^T + tr_b) ----------------
  {
    v16h ta[8];
#pragma unroll
    for (int kt = 0; kt < 8; ++kt) ta[kt] = a_from_f32(resid, 256, kt * 32, lane);
#pragma unroll
    for (int nt = 0; nt < 8; ++nt) {
      v8f c = bias8(P.tr_b, nt * 16 + nlo);
#pragma unroll
      for (int kt = 0; kt < 8; ++kt)
        c = wmma16(ta[kt], b_from_f16(P.wh + OFF_TR, 256, nt * 16, kt * 32, lane), c);
#pragma unroll
      for (int j = 0; j < 8; ++j) c[j] = fmaxf(c[j], 0.0f);
      store_c_f32(hbuf, HH, nt * 16, lane, c);
    }
  }
  __syncthreads();

  // ---------------- Phase H: heads ----------------
  if (lane < 16) {
    int r = lane; size_t b = (size_t)b0 + r;
    const float* hr = hbuf + (size_t)r * HH;
    float sm = P.hm_b[0], ss = P.hs_b[0], sd0 = P.hd_b[0], sd1 = P.hd_b[1];
    for (int c = 0; c < HH; ++c) {
      float x = hr[c];
      sm  += P.hm_w[c] * x;
      ss  += P.hs_w[c] * x;
      sd0 += P.hd_w[c] * x;
      sd1 += P.hd_w[HH + c] * x;
    }
    P.o_mean[b]       = sm;
    P.o_ls[b]         = ss;
    P.o_dir[2 * b]     = sd0;
    P.o_dir[2 * b + 1] = sd1;
  }
}

extern "C" void kernel_launch(void* const* d_in, const int* in_sizes, int n_in,
                              void* d_out, int out_size, void* d_ws, size_t ws_size,
                              hipStream_t stream) {
  (void)in_sizes; (void)n_in; (void)out_size; (void)ws_size;
  _Float16* wh = (_Float16*)d_ws;

  cvt_weights_kernel<<<(WS_HALVES + 255) / 256, 256, 0, stream>>>(
      (const float*)d_in[3],  (const float*)d_in[5],  (const float*)d_in[7],
      (const float*)d_in[15], (const float*)d_in[17], (const float*)d_in[19],
      (const float*)d_in[23], wh);

  Params P;
  P.x_seq = (const float*)d_in[0];
  P.aux   = (const float*)d_in[1];
  P.levy  = (const float*)d_in[2];
  P.ds_b  = (const float*)d_in[4];
  P.dr_b  = (const float*)d_in[6];
  P.df_b1 = (const float*)d_in[8];
  P.df_w2 = (const float*)d_in[9];
  P.df_b2 = (const float*)d_in[10];
  P.g_w1  = (const float*)d_in[11];
  P.g_b1  = (const float*)d_in[12];
  P.g_w2  = (const float*)d_in[13];
  P.g_b2  = (const float*)d_in[14];
  P.fs_b  = (const float*)d_in[16];
  P.in_b  = (const float*)d_in[18];
  P.out_b = (const float*)d_in[20];
  P.ln_g  = (const float*)d_in[21];
  P.ln_b  = (const float*)d_in[22];
  P.tr_b  = (const float*)d_in[24];
  P.hm_w  = (const float*)d_in[25];
  P.hm_b  = (const float*)d_in[26];
  P.hs_w  = (const float*)d_in[27];
  P.hs_b  = (const float*)d_in[28];
  P.hd_w  = (const float*)d_in[29];
  P.hd_b  = (const float*)d_in[30];
  P.wh    = wh;
  float* out = (float*)d_out;
  P.o_mean = out;
  P.o_ls   = out + BATCH;
  P.o_dir  = out + 2 * BATCH;

  DualStreamSDENet_fused_kernel<<<BATCH / 64, 128, WG_LDS, stream>>>(P);
}
